// STC_layer_58385785422536
// MI455X (gfx1250) — matrix operand order repaired and checked by reading, not compiled, not run on hardware
//
#include <hip/hip_runtime.h>
#include <hip/hip_bf16.h>

typedef unsigned int   u32;
typedef unsigned short u16;

typedef __bf16 v16bf __attribute__((ext_vector_type(16)));
typedef float  v8f   __attribute__((ext_vector_type(8)));

union BfFrag { u32 u[8]; v16bf v; };

#define NN    100000   // node universe
#define FEAT  512
#define HID   512
#define KNB   25       // neighbors per central node
#define FS    27       // filter size
#define NB    20000    // batch of central nodes
#define ALPHAS 0.2f

#define LDA 34         // LDS row pitch in halves (17 dwords, odd -> conflict-free)
#define LDB 34

__device__ __forceinline__ u16 f2bf(float x) {
  u32 u = __builtin_bit_cast(u32, x);
  u32 r = u + 0x7FFFu + ((u >> 16) & 1u);   // round-to-nearest-even
  return (u16)(r >> 16);
}
__device__ __forceinline__ u32 f2bf2(float lo, float hi) {
  return ((u32)f2bf(hi) << 16) | (u32)f2bf(lo);
}
__device__ __forceinline__ float bf2f(u16 h) {
  u32 t = ((u32)h) << 16;
  return __builtin_bit_cast(float, t);
}

// ---------------------------------------------------------------------------
// Kernel D: v = U @ (U^T @ ones)  (27-vector). Exact spectral collapse of
// ((mt @ U) @ U.T) @ ones without assuming U is perfectly orthogonal.
// ---------------------------------------------------------------------------
__global__ void spec_kernel(const float* __restrict__ U, float* __restrict__ v) {
  __shared__ float cs[FS];
  int t = threadIdx.x;
  if (t < FS) {
    float s = 0.f;
    for (int f = 0; f < FS; ++f) s += U[f * FS + t];   // column sums of U
    cs[t] = s;
  }
  __syncthreads();
  if (t < FS) {
    float a = 0.f;
    for (int j = 0; j < FS; ++j) a += U[t * FS + j] * cs[j];
    v[t] = a;
  }
}

// ---------------------------------------------------------------------------
// Kernel W: one-time W transpose + f32->bf16: Wt[n][k] = bf16(W[k][n]).
// ---------------------------------------------------------------------------
__global__ __launch_bounds__(256) void wprep_kernel(const float* __restrict__ W,
                                                    u16* __restrict__ Wt) {
  const int n = blockIdx.x;                 // 512 blocks: one output row each
  for (int k = threadIdx.x; k < FEAT; k += 256)
    Wt[n * FEAT + k] = f2bf(W[(size_t)k * HID + n]);
}

// ---------------------------------------------------------------------------
// Kernel A: H = feat_table @ W (bf16 out) via v_wmma_f32_16x16x32_bf16.
// 128x128 tile / 256-thread block; 8 waves in 4x2: each wave 32Mx64N
// (2 A frags x 4 B frags preloaded, 8 back-to-back WMMAs). 3-stage pipeline:
//   global loads (next slice) -> frag ds_loads + WMMAs (current) ->
//   ds_stores (next slice) -> barrier.  Double-buffered LDS.
// ---------------------------------------------------------------------------
__global__ __launch_bounds__(256) void proj_gemm(const float* __restrict__ X,
                                                 const u16* __restrict__ Wt,
                                                 u16* __restrict__ Hbf) {
  __shared__ u16 Asm[2][128 * LDA];   // A tile, row-major [m][k], bf16
  __shared__ u16 Bsm[2][128 * LDB];   // W tile [n][k], bf16

  const int tid    = threadIdx.x;
  const int wave   = tid >> 5;
  const int lane   = tid & 31;
  const int hl     = lane >> 4;         // half-wave selector
  const int r      = lane & 15;
  const int mw     = (wave >> 1) * 32;  // wave M offset in tile
  const int nw     = (wave & 1) * 64;   // wave N offset in tile
  const int m_base = blockIdx.x * 128;
  const int n0     = blockIdx.y * 128;

  // ---- hoisted staging addresses (constant across the K loop) ----
  const float* Xrow[4];
  int AsmOff[4];
  {
    const int q = (tid & 7) * 4;
#pragma unroll
    for (int p = 0; p < 4; ++p) {
      int row = p * 32 + (tid >> 3);
      int rg  = m_base + row;
      if (rg >= NN) rg = NN - 1;        // clamp; results discarded at store
      Xrow[p]   = X + (size_t)rg * FEAT + q;
      AsmOff[p] = row * LDA + q;
    }
  }
  const u16* Wsrc   = Wt + (size_t)(n0 + (tid >> 1)) * FEAT + (tid & 1) * 16;
  const int  BsmOff = (tid >> 1) * LDB + (tid & 1) * 16;

  const v8f zero = {0.f, 0.f, 0.f, 0.f, 0.f, 0.f, 0.f, 0.f};
  v8f acc[2][4];
#pragma unroll
  for (int ma = 0; ma < 2; ++ma)
#pragma unroll
    for (int nt = 0; nt < 4; ++nt) acc[ma][nt] = zero;

  // staging registers (next K-slice held in VGPRs between load and store)
  float4 sa[4];
  uint4  sb0, sb1;

  auto stage_load = [&](int kb) {       // global -> regs (issue early)
#pragma unroll
    for (int p = 0; p < 4; ++p) sa[p] = *(const float4*)(Xrow[p] + kb);
    const uint4* s = (const uint4*)(Wsrc + kb);
    sb0 = s[0];
    sb1 = s[1];
  };

  auto stage_store = [&](int buf) {     // regs -> LDS (after WMMAs)
#pragma unroll
    for (int p = 0; p < 4; ++p) {
      u32* dst = (u32*)&Asm[buf][AsmOff[p]];
      dst[0] = f2bf2(sa[p].x, sa[p].y);
      dst[1] = f2bf2(sa[p].z, sa[p].w);
    }
    u32* dst = (u32*)&Bsm[buf][BsmOff];
    dst[0] = sb0.x; dst[1] = sb0.y; dst[2] = sb0.z; dst[3] = sb0.w;
    dst[4] = sb1.x; dst[5] = sb1.y; dst[6] = sb1.z; dst[7] = sb1.w;
  };

  auto compute = [&](int buf) {
    // A frags, ISA 16-bit 16x32 layout:
    // lanes 0-15: K 0..7 in v0..3, K 16..23 in v4..7; lanes 16-31: +8 / +24
    BfFrag a[2];
#pragma unroll
    for (int ma = 0; ma < 2; ++ma) {
      const u16* Arow = &Asm[buf][(mw + ma * 16 + r) * LDA];
#pragma unroll
      for (int i = 0; i < 4; ++i) a[ma].u[i]     = *(const u32*)(Arow + hl * 8 + 2 * i);
#pragma unroll
      for (int i = 0; i < 4; ++i) a[ma].u[4 + i] = *(const u32*)(Arow + 16 + hl * 8 + 2 * i);
    }
    // All 4 B frags in distinct regs (avoids WMMA->ds_load WAR hazard NOPs)
    BfFrag b[4];
#pragma unroll
    for (int nt = 0; nt < 4; ++nt) {
      const u16* Brow = &Bsm[buf][(nw + nt * 16 + r) * LDB];
#pragma unroll
      for (int i = 0; i < 8; ++i) b[nt].u[i] = *(const u32*)(Brow + (i >> 2) * 16 + hl * 8 + 2 * (i & 3));
    }
    // 8 back-to-back WMMAs, no intervening VALU
#pragma unroll
    for (int nt = 0; nt < 4; ++nt) {
      acc[0][nt] = __builtin_amdgcn_wmma_f32_16x16x32_bf16(
          false, a[0].v, false, b[nt].v, (short)0, acc[0][nt], false, false);
      acc[1][nt] = __builtin_amdgcn_wmma_f32_16x16x32_bf16(
          false, a[1].v, false, b[nt].v, (short)0, acc[1][nt], false, false);
    }
  };

  stage_load(0);
  stage_store(0);
  __syncthreads();
  int cur = 0;
  for (int kb = 0; kb < FEAT; kb += 32) {
    const bool more = (kb + 32 < FEAT);
    if (more) stage_load(kb + 32);      // long-latency global loads first
    compute(cur);                       // frag ds_loads + 8 WMMAs
    if (more) stage_store(cur ^ 1);     // LDS fill of next buffer
    __syncthreads();
    cur ^= 1;
  }

  // store C per ISA 16x16 f32 layout: vgpr i -> M = hl*8 + i, N = r
#pragma unroll
  for (int ma = 0; ma < 2; ++ma)
#pragma unroll
    for (int nt = 0; nt < 4; ++nt) {
      int col = n0 + nw + nt * 16 + r;
#pragma unroll
      for (int i = 0; i < 8; ++i) {
        int row = m_base + mw + ma * 16 + hl * 8 + i;
        if (row < NN) Hbf[(size_t)row * HID + col] = f2bf(acc[ma][nt][i]);
      }
    }
}

// ---------------------------------------------------------------------------
// Kernel B: per-node attention scores s1 = H·a[:512], s2 = H·a[512:].
// One wave per row (wave32), shuffle reduction.
// ---------------------------------------------------------------------------
__global__ __launch_bounds__(256) void score_kernel(const u16* __restrict__ Hbf,
                                                    const float* __restrict__ a_vec,
                                                    float* __restrict__ S1,
                                                    float* __restrict__ S2) {
  const int wave = threadIdx.x >> 5;
  const int lane = threadIdx.x & 31;
  const int row  = blockIdx.x * 8 + wave;   // NN % 8 == 0
  const u16* hrow = Hbf + (size_t)row * HID;
  float d1 = 0.f, d2 = 0.f;
#pragma unroll
  for (int part = 0; part < 4; ++part) {
    int c  = lane * 16 + part * 4;
    u32 w0 = *(const u32*)(hrow + c);
    u32 w1 = *(const u32*)(hrow + c + 2);
    float f0 = bf2f((u16)(w0 & 0xffffu)), f1 = bf2f((u16)(w0 >> 16));
    float f2 = bf2f((u16)(w1 & 0xffffu)), f3 = bf2f((u16)(w1 >> 16));
    d1 += f0 * a_vec[c] + f1 * a_vec[c + 1] + f2 * a_vec[c + 2] + f3 * a_vec[c + 3];
    d2 += f0 * a_vec[HID + c]     + f1 * a_vec[HID + c + 1]
        + f2 * a_vec[HID + c + 2] + f3 * a_vec[HID + c + 3];
  }
#pragma unroll
  for (int off = 16; off >= 1; off >>= 1) {
    d1 += __shfl_xor(d1, off);
    d2 += __shfl_xor(d2, off);
  }
  if (lane == 0) { S1[row] = d1; S2[row] = d2; }
}

// ---------------------------------------------------------------------------
// Kernel C: per central node -> leaky-relu / exp(-.) attention over 25
// neighbors, fold in spectral weight v[k+1], gather-weighted sum of H rows,
// relu. One 128-thread block per node; 4 output columns per thread.
// ---------------------------------------------------------------------------
__global__ __launch_bounds__(128) void aggregate_kernel(const int* __restrict__ nodes,
                                                        const int* __restrict__ neigh,
                                                        const u16* __restrict__ Hbf,
                                                        const float* __restrict__ S1,
                                                        const float* __restrict__ S2,
                                                        const float* __restrict__ vspec,
                                                        float* __restrict__ out) {
  const int b = blockIdx.x;
  const float sc = S1[nodes[b]];
  int   nk[KNB];
  float wk[KNB];
  float sum = 0.f;
#pragma unroll
  for (int k = 0; k < KNB; ++k) {
    int n  = neigh[b * KNB + k];
    nk[k]  = n;
    float s  = sc + S2[n];
    float lr = s > 0.f ? s : ALPHAS * s;
    float e  = expf(-lr);
    wk[k] = e;
    sum  += e;
  }
  const float inv = 1.f / sum;
  const int c0 = threadIdx.x * 4;
  float a0 = 0.f, a1 = 0.f, a2 = 0.f, a3 = 0.f;
#pragma unroll
  for (int k = 0; k < KNB; ++k) {
    float w = wk[k] * inv * vspec[k + 1];    // pad slots 0 and 26 are zero
    const u16* hp = Hbf + (size_t)nk[k] * HID + c0;
    u32 q0 = *(const u32*)hp;
    u32 q1 = *(const u32*)(hp + 2);
    a0 += w * bf2f((u16)(q0 & 0xffffu));
    a1 += w * bf2f((u16)(q0 >> 16));
    a2 += w * bf2f((u16)(q1 & 0xffffu));
    a3 += w * bf2f((u16)(q1 >> 16));
  }
  float* op = out + (size_t)b * HID + c0;
  op[0] = fmaxf(a0, 0.f);
  op[1] = fmaxf(a1, 0.f);
  op[2] = fmaxf(a2, 0.f);
  op[3] = fmaxf(a3, 0.f);
}

// ---------------------------------------------------------------------------
extern "C" void kernel_launch(void* const* d_in, const int* in_sizes, int n_in,
                              void* d_out, int out_size, void* d_ws, size_t ws_size,
                              hipStream_t stream) {
  (void)in_sizes; (void)n_in; (void)out_size; (void)ws_size;
  const int*   nodes = (const int*)d_in[0];
  const int*   neigh = (const int*)d_in[1];
  const float* feat  = (const float*)d_in[2];
  const float* W     = (const float*)d_in[3];
  const float* a_vec = (const float*)d_in[4];
  const float* U     = (const float*)d_in[5];
  float* out = (float*)d_out;

  // workspace layout: H (bf16, 102.4MB) | S1 | S2 | v(pad 32) | Wt (bf16, 512KB)
  char* ws = (char*)d_ws;
  u16*  Hbf = (u16*)ws;
  size_t offH = (size_t)NN * HID * sizeof(u16);
  float* S1 = (float*)(ws + offH);
  float* S2 = S1 + NN;
  float* vs = S2 + NN;
  u16*   Wt = (u16*)(vs + 32);          // 16B-aligned

  spec_kernel<<<1, 32, 0, stream>>>(U, vs);
  wprep_kernel<<<HID, 256, 0, stream>>>(W, Wt);

  dim3 gA((NN + 127) / 128, HID / 128);
  proj_gemm<<<gA, dim3(256), 0, stream>>>(feat, Wt, Hbf);

  score_kernel<<<NN / 8, 256, 0, stream>>>(Hbf, a_vec, S1, S2);

  aggregate_kernel<<<NB, 128, 0, stream>>>(nodes, neigh, Hbf, S1, S2, vs, out);
}
